// TransformerALiBi_53472342835527
// MI455X (gfx1250) — matrix operand order repaired and checked by reading, not compiled, not run on hardware
//
#include <hip/hip_runtime.h>
#include <hip/hip_bf16.h>
#include <math.h>

// ---------------------------------------------------------------------------
// Types for CDNA5 WMMA (wave32): v16bf = A/B fragment, v8f = C/D fragment
// ---------------------------------------------------------------------------
typedef __bf16 v16bf __attribute__((ext_vector_type(16)));
typedef float  v8f   __attribute__((ext_vector_type(8)));

union FragBF { uint4 q[2]; v16bf v; };

__device__ inline v8f v8f_zero() {
    v8f z = {0.f,0.f,0.f,0.f,0.f,0.f,0.f,0.f};
    return z;
}

// LDS byte offset of a __shared__ object = low 32 bits of its generic address
// (aperture check uses addr[63:32]; ISA section 10.2)
__device__ inline unsigned lds_off(const void* p) {
    return (unsigned)(unsigned long long)(uintptr_t)p;
}

// Async DMA: global memory -> LDS, 16 bytes per lane, tracked by ASYNCcnt
__device__ inline void async_load_b128(unsigned lds, const void* gaddr) {
    asm volatile("global_load_async_to_lds_b128 %0, %1, off"
                 :: "v"(lds), "v"(gaddr) : "memory");
}
__device__ inline void wait_asynccnt0() {
    asm volatile("s_wait_asynccnt 0x0" ::: "memory");
}

// ---------------------------------------------------------------------------
// fp32 -> bf16 conversion
// ---------------------------------------------------------------------------
__global__ __launch_bounds__(256) void cvt_f32_bf16_kernel(const float* __restrict__ in,
                                                           __hip_bfloat16* __restrict__ out,
                                                           long n) {
    long i = (long)blockIdx.x * blockDim.x + threadIdx.x;
    long stride = (long)gridDim.x * blockDim.x;
    for (; i < n; i += stride) out[i] = __float2bfloat16(in[i]);
}

// ---------------------------------------------------------------------------
// Generic WMMA GEMM:  C[M,N] = A[M,K] * B^T  (B stored [N,K] when !TRB,
//                     B stored [K,N] when TRB — staged transposed via LDS)
// Double-buffered LDS; A (and non-transposed B) staged with
// GLOBAL_LOAD_ASYNC_TO_LDS_B128 overlapping the WMMA compute of the
// previous tile. One barrier + one s_wait_asynccnt per K-tile.
// Batched over blockIdx.z with (b, h) = (z / zdiv, z % zdiv) offsets.
// ---------------------------------------------------------------------------
template<int BM,int BN,int BK,int WM,int WN,int MT,int NT2,bool TRB,bool OUTF32,bool RELU>
__global__ __launch_bounds__(WM*WN*32)
void gemm_wmma(const unsigned short* __restrict__ Ag, long lda, long aOffB, long aOffH,
               const unsigned short* __restrict__ Bg, long ldb, long bOffB, long bOffH,
               const float* __restrict__ bias,
               void* __restrict__ Cg, long ldc, long cOffB, long cOffH,
               int K, int zdiv)
{
    constexpr int NTHREADS = WM*WN*32;
    constexpr int BKP = BK + 8;               // padded LDS row (16B-multiple stride)
    constexpr int KC  = BK/8;                 // 16B chunks per K row
    constexpr int NC  = BN/8;                 // 16B chunks per N row (TRB)
    constexpr int ACH = (BM*BK/8)/NTHREADS;   // A staging chunks per thread
    constexpr int BCH = ((TRB ? BK*BN : BN*BK)/8)/NTHREADS;
    static_assert((BM*BK/8) % NTHREADS == 0, "A staging divisibility");
    static_assert(((TRB ? BK*BN : BN*BK)/8) % NTHREADS == 0, "B staging divisibility");
    static_assert(BM == WM*MT*16 && BN == WN*NT2*16, "tile mismatch");

    __shared__ unsigned short Al[2][BM*BKP];
    __shared__ unsigned short Bl[2][BN*BKP];

    const int z  = blockIdx.z;
    const int zb = z / zdiv;
    const int zh = z % zdiv;
    Ag += zb*aOffB + zh*aOffH;
    Bg += zb*bOffB + zh*bOffH;
    char* Cbase = (char*)Cg + (zb*cOffB + zh*cOffH) * (long)(OUTF32 ? 4 : 2);

    const int tid  = threadIdx.x;
    const int wave = tid >> 5;
    const int lane = tid & 31;
    const int wm = wave / WN, wn = wave % WN;
    const int lm = lane & 15, lh = lane >> 4;

    const long blockM = (long)blockIdx.y * BM;
    const long blockN = (long)blockIdx.x * BN;

    // per-thread staging coordinates (compile-time chunk counts, no divergence)
    // async DMA staging of A tile and (if !TRB) B tile into buffer `buf`
    auto stage_async = [&](int buf, int k0) {
#pragma unroll
        for (int i = 0; i < ACH; ++i) {
            int c = tid + i*NTHREADS;
            int r = c / KC, q = c % KC;
            async_load_b128(lds_off(&Al[buf][r*BKP + q*8]),
                            Ag + (blockM + r)*lda + k0 + q*8);
        }
        if (!TRB) {
#pragma unroll
            for (int i = 0; i < BCH; ++i) {
                int c = tid + i*NTHREADS;
                int r = c / KC, q = c % KC;
                async_load_b128(lds_off(&Bl[buf][r*BKP + q*8]),
                                Bg + (blockN + r)*ldb + k0 + q*8);
            }
        }
    };
    // TRB: global load B [K,N] rows into registers
    auto trb_load = [&](int k0, uint4* tmp) {
        if (TRB) {
#pragma unroll
            for (int i = 0; i < BCH; ++i) {
                int c = tid + i*NTHREADS;
                int kr = c / NC, q = c % NC;
                tmp[i] = *(const uint4*)(Bg + (long)(k0 + kr)*ldb + blockN + q*8);
            }
        }
    };
    // TRB: scatter registers into LDS transposed as [n][k]
    auto trb_store = [&](int buf, const uint4* tmp) {
        if (TRB) {
#pragma unroll
            for (int i = 0; i < BCH; ++i) {
                int c = tid + i*NTHREADS;
                int kr = c / NC, q = c % NC;
                union { uint4 u; unsigned short s[8]; } d;
                d.u = tmp[i];
#pragma unroll
                for (int j = 0; j < 8; ++j) Bl[buf][(q*8 + j)*BKP + kr] = d.s[j];
            }
        }
    };

    v8f acc[MT][NT2];
#pragma unroll
    for (int i = 0; i < MT; ++i)
#pragma unroll
        for (int j = 0; j < NT2; ++j) acc[i][j] = v8f_zero();

    // prologue: stage first tile
    {
        stage_async(0, 0);
        uint4 t0[TRB ? BCH : 1];
        trb_load(0, t0);
        trb_store(0, t0);
    }

    int cur = 0;
    for (int k0 = 0; k0 < K; k0 += BK) {
        wait_asynccnt0();        // current buffer's DMA complete
        __syncthreads();         // all waves' staging visible

        const bool havenext = (k0 + BK) < K;
        uint4 tmpB[TRB ? BCH : 1];
        if (havenext) {
            stage_async(cur ^ 1, k0 + BK);   // DMA next tile during compute
            trb_load(k0 + BK, tmpB);         // global loads fly under WMMAs
        }

        // ---- compute on buffer `cur`: 16x16x32 k-steps ----
#pragma unroll
        for (int ks = 0; ks < BK; ks += 32) {
            // A 16x32 bf16: lane m=lm; a[0..7]=K[lh*8 .. +7], a[8..15]=K[16+lh*8 .. +7]
            FragBF a[MT], b[NT2];
#pragma unroll
            for (int i = 0; i < MT; ++i) {
                int m = wm*MT*16 + i*16 + lm;
                a[i].q[0] = *(const uint4*)&Al[cur][m*BKP + ks + lh*8];
                a[i].q[1] = *(const uint4*)&Al[cur][m*BKP + ks + 16 + lh*8];
            }
            // B 32x16 bf16: lane n=lm; b[0..7]=K[lh*8 .. +7], b[8..15]=K[16+lh*8 .. +7] of column n
#pragma unroll
            for (int j = 0; j < NT2; ++j) {
                int n = wn*NT2*16 + j*16 + lm;
                b[j].q[0] = *(const uint4*)&Bl[cur][n*BKP + ks + lh*8];
                b[j].q[1] = *(const uint4*)&Bl[cur][n*BKP + ks + 16 + lh*8];
            }
#pragma unroll
            for (int i = 0; i < MT; ++i)
#pragma unroll
                for (int j = 0; j < NT2; ++j)
                    acc[i][j] = __builtin_amdgcn_wmma_f32_16x16x32_bf16(
                        false, a[i].v, false, b[j].v, (short)0, acc[i][j], false, false);
        }

        if (havenext) trb_store(cur ^ 1, tmpB);  // after compute's ds_loads
        cur ^= 1;
    }

    // ---- epilogue: C/D layout: VGPR v, lane -> M = v + 8*lh, N = lm ----
#pragma unroll
    for (int i = 0; i < MT; ++i) {
        long mbase = blockM + wm*MT*16 + i*16 + lh*8;
#pragma unroll
        for (int j = 0; j < NT2; ++j) {
            long n = blockN + wn*NT2*16 + j*16 + lm;
            float bv = bias ? bias[n] : 0.f;
#pragma unroll
            for (int v = 0; v < 8; ++v) {
                float val = acc[i][j][v] + bv;
                if (RELU) val = fmaxf(val, 0.f);
                long m = mbase + v;
                if (OUTF32) ((float*)Cbase)[m*ldc + n] = val;
                else ((__hip_bfloat16*)Cbase)[m*ldc + n] = __float2bfloat16(val);
            }
        }
    }
}

// ---------------------------------------------------------------------------
// Row softmax with ALiBi: P[row,s] = softmax_s( S[row,s]*scale - slope*|t-s| )
// rows = B*H*T, 1024 cols, 128 threads/block, 8 cols/thread
// ---------------------------------------------------------------------------
__global__ __launch_bounds__(128)
void softmax_alibi_kernel(const float* __restrict__ S, __hip_bfloat16* __restrict__ P,
                          float scale, float slope)
{
    __shared__ float red[128];
    const long row = blockIdx.x;
    const int  t   = (int)(row & 1023);
    const int  tid = threadIdx.x;
    const float* sr = S + row*1024;

    float v[8]; float mx = -1e30f;
#pragma unroll
    for (int j = 0; j < 8; ++j) {
        int s = j*128 + tid;
        float x = sr[s]*scale - slope*fabsf((float)(t - s));
        v[j] = x; mx = fmaxf(mx, x);
    }
    red[tid] = mx; __syncthreads();
    for (int o = 64; o > 0; o >>= 1) {
        if (tid < o) red[tid] = fmaxf(red[tid], red[tid+o]);
        __syncthreads();
    }
    mx = red[0]; __syncthreads();

    float sum = 0.f;
#pragma unroll
    for (int j = 0; j < 8; ++j) { v[j] = __expf(v[j]-mx); sum += v[j]; }
    red[tid] = sum; __syncthreads();
    for (int o = 64; o > 0; o >>= 1) {
        if (tid < o) red[tid] += red[tid+o];
        __syncthreads();
    }
    float inv = 1.f / red[0];
#pragma unroll
    for (int j = 0; j < 8; ++j)
        P[row*1024 + j*128 + tid] = __float2bfloat16(v[j]*inv);
}

// ---------------------------------------------------------------------------
// Fused residual + LayerNorm (in place on h): h = LN(h + y) * g + b
// rows = B*T, D=1024 cols, 128 threads/block
// ---------------------------------------------------------------------------
__global__ __launch_bounds__(128)
void add_ln_kernel(__hip_bfloat16* __restrict__ h, const __hip_bfloat16* __restrict__ y,
                   const float* __restrict__ g, const float* __restrict__ b)
{
    __shared__ float red[128];
    const long row = blockIdx.x;
    const int  tid = threadIdx.x;
    __hip_bfloat16* hr = h + row*1024;
    const __hip_bfloat16* yr = y + row*1024;

    float v[8]; float s = 0.f;
#pragma unroll
    for (int j = 0; j < 8; ++j) {
        int c = j*128 + tid;
        v[j] = __bfloat162float(hr[c]) + __bfloat162float(yr[c]);
        s += v[j];
    }
    red[tid] = s; __syncthreads();
    for (int o = 64; o > 0; o >>= 1) { if (tid < o) red[tid] += red[tid+o]; __syncthreads(); }
    float mu = red[0] * (1.f/1024.f); __syncthreads();

    float ss = 0.f;
#pragma unroll
    for (int j = 0; j < 8; ++j) { float d = v[j]-mu; ss += d*d; }
    red[tid] = ss; __syncthreads();
    for (int o = 64; o > 0; o >>= 1) { if (tid < o) red[tid] += red[tid+o]; __syncthreads(); }
    float rstd = rsqrtf(red[0] * (1.f/1024.f) + 1e-5f);

#pragma unroll
    for (int j = 0; j < 8; ++j) {
        int c = j*128 + tid;
        hr[c] = __float2bfloat16((v[j]-mu)*rstd*g[c] + b[c]);
    }
}

// ---------------------------------------------------------------------------
// Head: out[b,c] = dot(h[b, T-1, :], head_w[c,:]) + head_b[c]   (B*C blocks)
// ---------------------------------------------------------------------------
__global__ __launch_bounds__(256)
void head_kernel(const __hip_bfloat16* __restrict__ h, const float* __restrict__ w,
                 const float* __restrict__ bias, float* __restrict__ out)
{
    __shared__ float red[256];
    const int idx = blockIdx.x;
    const int b = idx / 10, c = idx % 10;
    const int tid = threadIdx.x;
    const __hip_bfloat16* hr = h + ((long)b*1024 + 1023)*1024;
    const float* wr = w + (long)c*1024;
    float s = 0.f;
    for (int i = tid; i < 1024; i += 256) s += __bfloat162float(hr[i]) * wr[i];
    red[tid] = s; __syncthreads();
    for (int o = 128; o > 0; o >>= 1) { if (tid < o) red[tid] += red[tid+o]; __syncthreads(); }
    if (tid == 0) out[idx] = red[0] + bias[c];
}

// ---------------------------------------------------------------------------
// Host launcher
// ---------------------------------------------------------------------------
extern "C" void kernel_launch(void* const* d_in, const int* in_sizes, int n_in,
                              void* d_out, int out_size, void* d_ws, size_t ws_size,
                              hipStream_t stream)
{
    (void)in_sizes; (void)n_in; (void)out_size; (void)ws_size;
    enum { Lc=6, Bc=4, Tc=1024, INc=256, Dc=1024, Hc=16, FFc=4096, Cc=10 };
    const long BT = (long)Bc*Tc;           // 4096

    const float* x       = (const float*)d_in[0];
    const float* input_w = (const float*)d_in[1];
    const float* input_b = (const float*)d_in[2];
    const float* Wqkv    = (const float*)d_in[3];
    const float* bqkv    = (const float*)d_in[4];
    const float* Wo      = (const float*)d_in[5];
    const float* bo      = (const float*)d_in[6];
    const float* ln1_g   = (const float*)d_in[7];
    const float* ln1_b   = (const float*)d_in[8];
    const float* ln2_g   = (const float*)d_in[9];
    const float* ln2_b   = (const float*)d_in[10];
    const float* W1      = (const float*)d_in[11];
    const float* b1      = (const float*)d_in[12];
    const float* W2      = (const float*)d_in[13];
    const float* b2      = (const float*)d_in[14];
    const float* head_w  = (const float*)d_in[15];
    const float* head_b  = (const float*)d_in[16];
    float* out = (float*)d_out;

    // ALiBi mean slope: slopes_i = 2^{-8 i / 16}, i=1..16
    float slope = 0.f;
    for (int i = 1; i <= 16; ++i) slope += powf(2.f, -8.f*(float)i/16.f);
    slope /= 16.f;
    const float scale = 0.125f;  // 1/sqrt(64)

    // workspace carve-out
    char* ws = (char*)d_ws; size_t off = 0;
    auto alloc = [&](size_t bytes) -> void* {
        void* p = ws + off; off = (off + bytes + 255) & ~(size_t)255; return p;
    };
    typedef __hip_bfloat16 bf;
    bf*    xb    = (bf*)   alloc(BT*INc*2);
    bf*    wIn   = (bf*)   alloc((size_t)Dc*INc*2);
    bf*    wQkv  = (bf*)   alloc((size_t)Lc*3*Dc*Dc*2);
    bf*    wO    = (bf*)   alloc((size_t)Lc*Dc*Dc*2);
    bf*    w1    = (bf*)   alloc((size_t)Lc*FFc*Dc*2);
    bf*    w2    = (bf*)   alloc((size_t)Lc*Dc*FFc*2);
    bf*    hb    = (bf*)   alloc(BT*Dc*2);
    bf*    qkvb  = (bf*)   alloc(BT*3*Dc*2);
    float* Sbuf  = (float*)alloc((size_t)Bc*Hc*Tc*Tc*4);
    bf*    Pbuf  = (bf*)   alloc((size_t)Bc*Hc*Tc*Tc*2);
    bf*    attnO = (bf*)   alloc(BT*Dc*2);
    bf*    oproj = (bf*)   alloc(BT*Dc*2);
    bf*    ff1b  = (bf*)   alloc(BT*FFc*2);
    bf*    ff2b  = (bf*)   alloc(BT*Dc*2);

    // fp32 -> bf16 conversions (x + all big weights)
    auto cvt = [&](const float* src, bf* dst, long n) {
        cvt_f32_bf16_kernel<<<dim3(2048), dim3(256), 0, stream>>>(src, dst, n);
    };
    cvt(x,       xb,   BT*INc);
    cvt(input_w, wIn,  (long)Dc*INc);
    cvt(Wqkv,    wQkv, (long)Lc*3*Dc*Dc);
    cvt(Wo,      wO,   (long)Lc*Dc*Dc);
    cvt(W1,      w1,   (long)Lc*FFc*Dc);
    cvt(W2,      w2,   (long)Lc*Dc*FFc);

    typedef const unsigned short* us;

    // input projection: h = x @ input_w^T + input_b     [4096 x 1024], K=256
    gemm_wmma<128,128,32,2,4,4,2,false,false,false>
        <<<dim3(Dc/128, BT/128, 1), 256, 0, stream>>>(
        (us)xb, INc, 0,0, (us)wIn, INc, 0,0, input_b,
        hb, Dc, 0,0, INc, 1);

    for (int l = 0; l < Lc; ++l) {
        // qkv = h @ Wqkv[l]^T + bqkv[l]   [4096 x 3072], K=1024
        gemm_wmma<128,128,32,2,4,4,2,false,false,false>
            <<<dim3(3*Dc/128, BT/128, 1), 256, 0, stream>>>(
            (us)hb, Dc, 0,0, (us)(wQkv + (long)l*3*Dc*Dc), Dc, 0,0,
            bqkv + (long)l*3*Dc, qkvb, 3*Dc, 0,0, Dc, 1);

        // scores = Q @ K^T per (b,h): batched NT, f32 out   [1024x1024], K=64
        gemm_wmma<128,128,32,2,4,4,2,false,true,false>
            <<<dim3(Tc/128, Tc/128, Bc*Hc), 256, 0, stream>>>(
            (us)qkvb,        3*Dc, (long)Tc*3*Dc, 64,
            (us)(qkvb + Dc), 3*Dc, (long)Tc*3*Dc, 64,
            nullptr,
            Sbuf, Tc, (long)Hc*Tc*Tc, (long)Tc*Tc, 64, Hc);

        // softmax with ALiBi -> P (bf16)
        softmax_alibi_kernel<<<dim3(Bc*Hc*Tc), 128, 0, stream>>>(Sbuf, Pbuf, scale, slope);

        // O = P @ V per (b,h): batched NN (B staged transposed)  [1024x64], K=1024
        gemm_wmma<128,64,32,4,2,2,2,true,false,false>
            <<<dim3(1, Tc/128, Bc*Hc), 256, 0, stream>>>(
            (us)Pbuf,           Tc,   (long)Hc*Tc*Tc, (long)Tc*Tc,
            (us)(qkvb + 2*Dc),  3*Dc, (long)Tc*3*Dc,  64,
            nullptr,
            attnO, Dc, (long)Tc*Dc, 64, Tc, Hc);

        // o = O @ Wo[l]^T + bo[l]   [4096 x 1024], K=1024
        gemm_wmma<128,128,32,2,4,4,2,false,false,false>
            <<<dim3(Dc/128, BT/128, 1), 256, 0, stream>>>(
            (us)attnO, Dc, 0,0, (us)(wO + (long)l*Dc*Dc), Dc, 0,0,
            bo + (long)l*Dc, oproj, Dc, 0,0, Dc, 1);

        // h = LN(h + o)
        add_ln_kernel<<<dim3((unsigned)BT), 128, 0, stream>>>(
            hb, oproj, ln1_g + (long)l*Dc, ln1_b + (long)l*Dc);

        // ff = relu(h @ W1[l]^T + b1[l])   [4096 x 4096], K=1024
        gemm_wmma<128,128,32,2,4,4,2,false,false,true>
            <<<dim3(FFc/128, BT/128, 1), 256, 0, stream>>>(
            (us)hb, Dc, 0,0, (us)(w1 + (long)l*FFc*Dc), Dc, 0,0,
            b1 + (long)l*FFc, ff1b, FFc, 0,0, Dc, 1);

        // ff2 = ff @ W2[l]^T + b2[l]   [4096 x 1024], K=4096
        gemm_wmma<128,128,32,2,4,4,2,false,false,false>
            <<<dim3(Dc/128, BT/128, 1), 256, 0, stream>>>(
            (us)ff1b, FFc, 0,0, (us)(w2 + (long)l*Dc*FFc), FFc, 0,0,
            b2 + (long)l*Dc, ff2b, Dc, 0,0, FFc, 1);

        // h = LN(h + ff2)
        add_ln_kernel<<<dim3((unsigned)BT), 128, 0, stream>>>(
            hb, ff2b, ln2_g + (long)l*Dc, ln2_b + (long)l*Dc);
    }

    // head: out[b,c] from last token
    head_kernel<<<dim3(Bc*Cc), 256, 0, stream>>>(hb, head_w, head_b, out);
}
